// GRU_Dropout_Model_44049184587869
// MI455X (gfx1250) — matrix operand order repaired and checked by reading, not compiled, not run on hardware
//
#include <hip/hip_runtime.h>
#include <hip/hip_bf16.h>
#include <math.h>

// ---------------------------------------------------------------------------
// Types for CDNA5 WMMA (gfx1250, wave32)
// ---------------------------------------------------------------------------
typedef __attribute__((ext_vector_type(16))) __bf16 bf16x16;
typedef __attribute__((ext_vector_type(8)))  float  floatx8;

__device__ __forceinline__ unsigned short f32_to_bf16_bits(float f) {
    unsigned int u = __float_as_uint(f);
    u += 0x7FFFu + ((u >> 16) & 1u);   // round-to-nearest-even
    return (unsigned short)(u >> 16);
}

__device__ __forceinline__ float fast_sigmoid(float x) {
    return __builtin_amdgcn_rcpf(1.0f + __expf(-x));        // v_exp + v_rcp
}
__device__ __forceinline__ float fast_tanh(float x) {
    return 1.0f - 2.0f * __builtin_amdgcn_rcpf(__expf(2.0f * x) + 1.0f);
}

// ---------------------------------------------------------------------------
// Weight repack: fp32 (Ksrc x N, row-major) -> bf16 packed WMMA-B fragments.
// Tile (kt, nt) of 32x16 stored as 32 lanes x 16 bf16 contiguous:
//   lane L: column n = nt*16 + (L%16), elements e=0..15 -> k = kt*32 + (L/16)*16 + e
// ---------------------------------------------------------------------------
__global__ void pack_weight_bf16(const float* __restrict__ W,
                                 unsigned short* __restrict__ pk,
                                 int Ksrc, int Kpad, int N) {
    int i = blockIdx.x * blockDim.x + threadIdx.x;
    if (i >= Kpad * N) return;
    int e    = i & 15;
    int lane = (i >> 4) & 31;
    int nt   = (i >> 9) % (N >> 4);
    int kt   = i / ((N >> 4) << 9);
    int n    = (nt << 4) + (lane & 15);
    int k    = (kt << 5) + ((lane >> 4) << 4) + e;
    float v  = (k < Ksrc) ? W[(size_t)k * N + n] : 0.0f;
    pk[i] = f32_to_bf16_bits(v);
}

__global__ void zero_f32(float* __restrict__ p, int n) {
    int i = blockIdx.x * blockDim.x + threadIdx.x;
    if (i < n) p[i] = 0.0f;
}

// ---------------------------------------------------------------------------
// Issue one 64x32 fp32 A-tile into LDS via GLOBAL_LOAD_ASYNC_TO_LDS_B128.
// A = concat(xt[64 x din], second[64 x dh], zero-pad). Every 16B chunk lies
// entirely in one source (boundaries din and din+dh are multiples of 4).
// ---------------------------------------------------------------------------
__device__ __forceinline__ void issue_async_tile(
        const float* xt, const float* second, const float* zbuf,
        int din, int dh, int m_base, int k0, unsigned dst_base, int tid) {
    #pragma unroll
    for (int i = 0; i < 4; ++i) {      // 512 16B chunks / 128 threads
        int q  = tid + i * 128;
        int r  = q >> 3, c4 = (q & 7) << 2;
        int gk = k0 + c4, gm = m_base + r;
        const float* src;
        if (gk < din) src = xt + (size_t)gm * din + gk;
        else { int kk = gk - din;
               src = (kk < dh) ? second + (size_t)gm * dh + kk : zbuf; }
        unsigned dst = dst_base + (unsigned)(q << 4);
        asm volatile("global_load_async_to_lds_b128 %0, %1, off"
                     :: "v"(dst), "v"(src) : "memory");
    }
}

// ---------------------------------------------------------------------------
// Shared GEMM core: C(64x64 block) = concat(xt, second)[64 x Kpad] @ Wpk.
// 128 threads = 4 waves; wave w computes rows [w*16, w*16+16) x 64 cols.
// Per K-step schedule (latency-oriented, low-occupancy-safe):
//   1) issue next A-tile asyncs into the other LDS buffer (ASYNCcnt path)
//   2) issue ALL 8 global_load_b128 for this step's B fragments  <-- before
//   3) s_wait_asynccnt + barrier                                     the sync,
//   4) A fragment: 4x ds_load_b128 + v_cvt_pk_bf16_f32                so B's L2
//   5) 4x v_wmma_f32_16x16x32_bf16                                    latency is
//   6) barrier                                                        hidden
// ---------------------------------------------------------------------------
#define GRU_GEMM_CORE(SECOND_PTR)                                              \
    __shared__ __align__(16) float ldsAf[2][64 * 32];                          \
    const int m_base = blockIdx.x * 64;                                        \
    const int n_base = blockIdx.y * 64;                                        \
    const int wave = threadIdx.x >> 5;                                         \
    const int lane = threadIdx.x & 31;                                         \
    const int lmod = lane & 15;                                                \
    const int half = lane >> 4;                                                \
    const unsigned lds0 = (unsigned)(size_t)(const void*)&ldsAf[0][0];         \
    const unsigned lds1 = (unsigned)(size_t)(const void*)&ldsAf[1][0];         \
    floatx8 acc[4] = {};                                                       \
    const int ntiles = N >> 4;                                                 \
    const int nsteps = Kpad >> 5;                                              \
    issue_async_tile(xt, (SECOND_PTR), zbuf, din, dh, m_base, 0, lds0,         \
                     threadIdx.x);                                             \
    for (int s = 0; s < nsteps; ++s) {                                         \
        const int k0 = s << 5;                                                 \
        if (s + 1 < nsteps)                                                    \
            issue_async_tile(xt, (SECOND_PTR), zbuf, din, dh, m_base,          \
                             k0 + 32, (s & 1) ? lds0 : lds1, threadIdx.x);     \
        /* B fragments: issued BEFORE the A-tile sync so L2 latency overlaps */\
        const unsigned short* bbase =                                          \
            Wpk + ((size_t)(k0 >> 5) * ntiles + (n_base >> 4)) * 512 + lane * 16;\
        bf16x16 bfr[4];                                                        \
        _Pragma("unroll")                                                      \
        for (int j = 0; j < 4; ++j) {                                          \
            const unsigned short* bp = bbase + j * 512;                        \
            ((uint4*)&bfr[j])[0] = ((const uint4*)bp)[0];                      \
            ((uint4*)&bfr[j])[1] = ((const uint4*)bp)[1];                      \
        }                                                                      \
        __builtin_prefetch(bbase + (size_t)ntiles * 512, 0, 1);                \
        if (s + 1 < nsteps)                                                    \
            asm volatile("s_wait_asynccnt 0x4" ::: "memory");                  \
        else                                                                   \
            asm volatile("s_wait_asynccnt 0x0" ::: "memory");                  \
        __syncthreads();                                                       \
        const float* abuf = (s & 1) ? &ldsAf[1][0] : &ldsAf[0][0];             \
        /* A layout: lanes0-15 K[0-15]; lanes16-31 K[16-31] */                 \
        bf16x16 afrag;                                                         \
        {                                                                      \
            const float* arow = abuf + (wave * 16 + lmod) * 32 + half * 16;    \
            float4 fa = ((const float4*)arow)[0];                              \
            float4 fb = ((const float4*)arow)[1];                              \
            float4 fc = ((const float4*)(arow + 8))[0];                        \
            float4 fd = ((const float4*)(arow + 8))[1];                        \
            afrag[0]  = (__bf16)fa.x; afrag[1]  = (__bf16)fa.y;                \
            afrag[2]  = (__bf16)fa.z; afrag[3]  = (__bf16)fa.w;                \
            afrag[4]  = (__bf16)fb.x; afrag[5]  = (__bf16)fb.y;                \
            afrag[6]  = (__bf16)fb.z; afrag[7]  = (__bf16)fb.w;                \
            afrag[8]  = (__bf16)fc.x; afrag[9]  = (__bf16)fc.y;                \
            afrag[10] = (__bf16)fc.z; afrag[11] = (__bf16)fc.w;                \
            afrag[12] = (__bf16)fd.x; afrag[13] = (__bf16)fd.y;                \
            afrag[14] = (__bf16)fd.z; afrag[15] = (__bf16)fd.w;                \
        }                                                                      \
        _Pragma("unroll")                                                      \
        for (int j = 0; j < 4; ++j) {                                          \
            acc[j] = __builtin_amdgcn_wmma_f32_16x16x32_bf16(                  \
                false, afrag, false, bfr[j], (short)0, acc[j], false, false);  \
        }                                                                      \
        __syncthreads();                                                       \
    }

// Gate kernel: ru = sigmoid(concat(xt,h)@Wg + bg); writes rh = r*h, u.
// dh % 64 == 0, so the r/u split is uniform per block (scalar branch).
__global__ __launch_bounds__(128)
void gru_gate_kernel(const float* __restrict__ xt, int din,
                     const float* __restrict__ h, int dh,
                     const unsigned short* __restrict__ Wpk, int Kpad, int N,
                     const float* __restrict__ bias, const float* __restrict__ zbuf,
                     float* __restrict__ rh, float* __restrict__ u_out) {
    GRU_GEMM_CORE(h)
    // D layout: VGPR v -> row = v + (half?8:0); col = lane%16
    if (n_base < dh) {
        #pragma unroll
        for (int j = 0; j < 4; ++j) {
            #pragma unroll
            for (int v = 0; v < 8; ++v) {
                int m = m_base + wave * 16 + v + (half ? 8 : 0);
                int n = n_base + j * 16 + lmod;
                float s = fast_sigmoid(acc[j][v] + bias[n]);
                rh[(size_t)m * dh + n] = s * h[(size_t)m * dh + n];
            }
        }
    } else {
        #pragma unroll
        for (int j = 0; j < 4; ++j) {
            #pragma unroll
            for (int v = 0; v < 8; ++v) {
                int m = m_base + wave * 16 + v + (half ? 8 : 0);
                int n = n_base + j * 16 + lmod;
                float s = fast_sigmoid(acc[j][v] + bias[n]);
                u_out[(size_t)m * dh + (n - dh)] = s;
            }
        }
    }
}

// Cell kernel: c = tanh(concat(xt,rh)@Wc + bc); h = u*h+(1-u)*c; y = h*mask.
__global__ __launch_bounds__(128)
void gru_cell_kernel(const float* __restrict__ xt, int din,
                     const float* __restrict__ rh, int dh,
                     const unsigned short* __restrict__ Wpk, int Kpad, int N,
                     const float* __restrict__ bias, const float* __restrict__ zbuf,
                     const float* __restrict__ u, float* __restrict__ h,
                     const float* __restrict__ mask_t, float* __restrict__ ybuf) {
    GRU_GEMM_CORE(rh)
    #pragma unroll
    for (int j = 0; j < 4; ++j) {
        #pragma unroll
        for (int v = 0; v < 8; ++v) {
            int m = m_base + wave * 16 + v + (half ? 8 : 0);
            int n = n_base + j * 16 + lmod;
            float c  = fast_tanh(acc[j][v] + bias[n]);
            size_t o = (size_t)m * dh + n;
            float uu = u[o];
            float hn = uu * h[o] + (1.0f - uu) * c;
            h[o]    = hn;
            ybuf[o] = hn * mask_t[o];
        }
    }
}

// Final L2 normalization of embedded = y3 (256 x 256).
__global__ void l2norm_rows(const float* __restrict__ e, float* __restrict__ out, int P) {
    __shared__ float red[256];
    int m = blockIdx.x, t = threadIdx.x;
    float v = e[(size_t)m * P + t];
    red[t] = v * v;
    __syncthreads();
    for (int s = 128; s > 0; s >>= 1) {
        if (t < s) red[t] += red[t + s];
        __syncthreads();
    }
    out[(size_t)m * P + t] = v / (sqrtf(red[0]) + 1e-6f);
}

// ---------------------------------------------------------------------------
// Host launcher
// ---------------------------------------------------------------------------
extern "C" void kernel_launch(void* const* d_in, const int* in_sizes, int n_in,
                              void* d_out, int out_size, void* d_ws, size_t ws_size,
                              hipStream_t stream) {
    (void)in_sizes; (void)n_in; (void)out_size; (void)ws_size;
    const int T = 160, B = 256, F = 40, H = 768, P = 256;

    const float* x    = (const float*)d_in[0];
    const float* Wg1  = (const float*)d_in[1];
    const float* bg1  = (const float*)d_in[2];
    const float* Wc1  = (const float*)d_in[3];
    const float* bc1  = (const float*)d_in[4];
    const float* Wg2  = (const float*)d_in[5];
    const float* bg2  = (const float*)d_in[6];
    const float* Wc2  = (const float*)d_in[7];
    const float* bc2  = (const float*)d_in[8];
    const float* Wg3  = (const float*)d_in[9];
    const float* bg3  = (const float*)d_in[10];
    const float* Wc3  = (const float*)d_in[11];
    const float* bc3  = (const float*)d_in[12];
    const float* msk1 = (const float*)d_in[13];
    const float* msk2 = (const float*)d_in[14];
    const float* msk3 = (const float*)d_in[15];

    // workspace carving (256B aligned)
    size_t off = 0;
    char* base = (char*)d_ws;
    auto take = [&](size_t bytes) -> char* {
        char* p = base + off;
        off += (bytes + 255) & ~(size_t)255;
        return p;
    };
    // padded K dims: L1: 40+768=808 -> 832; L2: 768+768=1536; L3: 768+256=1024
    unsigned short* Wg1p = (unsigned short*)take((size_t)832  * 1536 * 2);
    unsigned short* Wc1p = (unsigned short*)take((size_t)832  * 768  * 2);
    unsigned short* Wg2p = (unsigned short*)take((size_t)1536 * 1536 * 2);
    unsigned short* Wc2p = (unsigned short*)take((size_t)1536 * 768  * 2);
    unsigned short* Wg3p = (unsigned short*)take((size_t)1024 * 512  * 2);
    unsigned short* Wc3p = (unsigned short*)take((size_t)1024 * 256  * 2);
    float* h1  = (float*)take((size_t)B * H * 4);
    float* rh1 = (float*)take((size_t)B * H * 4);
    float* u1  = (float*)take((size_t)B * H * 4);
    float* y1  = (float*)take((size_t)B * H * 4);
    float* h2  = (float*)take((size_t)B * H * 4);
    float* rh2 = (float*)take((size_t)B * H * 4);
    float* u2  = (float*)take((size_t)B * H * 4);
    float* y2  = (float*)take((size_t)B * H * 4);
    float* h3  = (float*)take((size_t)B * P * 4);
    float* rh3 = (float*)take((size_t)B * P * 4);
    float* u3  = (float*)take((size_t)B * P * 4);
    float* y3  = (float*)take((size_t)B * P * 4);
    float* zbuf = (float*)take(64 * 4);   // zero source for async K-padding

    // one-time (per call) weight repack to bf16 WMMA-B fragments (L2 resident)
    auto packgrid = [](int kpad, int n) { return dim3((kpad * n + 255) / 256); };
    hipLaunchKernelGGL(pack_weight_bf16, packgrid(832, 1536), dim3(256), 0, stream, Wg1, Wg1p, 808, 832, 1536);
    hipLaunchKernelGGL(pack_weight_bf16, packgrid(832, 768),  dim3(256), 0, stream, Wc1, Wc1p, 808, 832, 768);
    hipLaunchKernelGGL(pack_weight_bf16, packgrid(1536,1536), dim3(256), 0, stream, Wg2, Wg2p, 1536, 1536, 1536);
    hipLaunchKernelGGL(pack_weight_bf16, packgrid(1536, 768), dim3(256), 0, stream, Wc2, Wc2p, 1536, 1536, 768);
    hipLaunchKernelGGL(pack_weight_bf16, packgrid(1024, 512), dim3(256), 0, stream, Wg3, Wg3p, 1024, 1024, 512);
    hipLaunchKernelGGL(pack_weight_bf16, packgrid(1024, 256), dim3(256), 0, stream, Wc3, Wc3p, 1024, 1024, 256);

    hipLaunchKernelGGL(zero_f32, dim3((B * H + 255) / 256), dim3(256), 0, stream, h1, B * H);
    hipLaunchKernelGGL(zero_f32, dim3((B * H + 255) / 256), dim3(256), 0, stream, h2, B * H);
    hipLaunchKernelGGL(zero_f32, dim3((B * P + 255) / 256), dim3(256), 0, stream, h3, B * P);
    hipLaunchKernelGGL(zero_f32, dim3(1), dim3(64), 0, stream, zbuf, 64);

    const dim3 blk(128);
    // grid.x = B/64 = 4 always; grid.y = N/64
    for (int t = 0; t < T; ++t) {
        const float* x1t = x + (size_t)t * B * F;
        // layer 1 (din=40, dh=768, Kpad=832)
        hipLaunchKernelGGL(gru_gate_kernel, dim3(4, 2 * H / 64), blk, 0, stream,
                           x1t, F, h1, H, Wg1p, 832, 2 * H, bg1, zbuf, rh1, u1);
        hipLaunchKernelGGL(gru_cell_kernel, dim3(4, H / 64), blk, 0, stream,
                           x1t, F, rh1, H, Wc1p, 832, H, bc1, zbuf, u1, h1,
                           msk1 + (size_t)t * B * H, y1);
        // layer 2 (din=768, dh=768, Kpad=1536)
        hipLaunchKernelGGL(gru_gate_kernel, dim3(4, 2 * H / 64), blk, 0, stream,
                           y1, H, h2, H, Wg2p, 1536, 2 * H, bg2, zbuf, rh2, u2);
        hipLaunchKernelGGL(gru_cell_kernel, dim3(4, H / 64), blk, 0, stream,
                           y1, H, rh2, H, Wc2p, 1536, H, bc2, zbuf, u2, h2,
                           msk2 + (size_t)t * B * H, y2);
        // layer 3 (din=768, dh=256, Kpad=1024)
        hipLaunchKernelGGL(gru_gate_kernel, dim3(4, 2 * P / 64), blk, 0, stream,
                           y2, H, h3, P, Wg3p, 1024, 2 * P, bg3, zbuf, rh3, u3);
        hipLaunchKernelGGL(gru_cell_kernel, dim3(4, P / 64), blk, 0, stream,
                           y2, H, rh3, P, Wc3p, 1024, P, bc3, zbuf, u3, h3,
                           msk3 + (size_t)t * B * P, y3);
    }
    // embedded = y3 (last step's masked h3); L2-normalize rows into d_out
    hipLaunchKernelGGL(l2norm_rows, dim3(B), dim3(P), 0, stream, y3, (float*)d_out, P);
}
